// GCN_31353261261178
// MI455X (gfx1250) — matrix operand order repaired and checked by reading, not compiled, not run on hardware
//
#include <hip/hip_runtime.h>
#include <hip/hip_bf16.h>

// GCN (2x GCNConv + LN + ReLU + mean-pool + linear) for MI455X / gfx1250.
//
// Strategy:
//  - Node GEMMs (x@W, 100000x128 @ 128x128) via V_WMMA_F32_16X16X32_BF16,
//    bf16 inputs / f32 accumulation. W is converted to bf16 and staged
//    TRANSPOSED in LDS so each B fragment is two contiguous 16B LDS reads.
//  - Edge message passing: one wave32 per edge; float4 gather of h[src],
//    native global_atomic_add_f32 scatter into agg[dst]. The 51MB feature
//    matrix is fully resident in the 192MB L2, so gather+atomics stay on-chip.
//  - Self-loop + bias + LayerNorm + ReLU fused into one row-per-wave kernel
//    (wave32 shuffle reductions); layer-2 variant also fuses graph mean-pool.

#define FEAT    128
#define NGRAPH  64
#define DOUT    16

typedef __attribute__((ext_vector_type(16))) __bf16 v16bf;
typedef __attribute__((ext_vector_type(16))) short  v16s;
typedef __attribute__((ext_vector_type(8)))  float  v8f;

__device__ __forceinline__ unsigned short f2bf(float f) {
  // round-to-nearest-even fp32 -> bf16
  unsigned u = __float_as_uint(f);
  u += 0x7FFFu + ((u >> 16) & 1u);
  return (unsigned short)(u >> 16);
}

__device__ __forceinline__ float wave_sum32(float v) {
#pragma unroll
  for (int o = 16; o > 0; o >>= 1) v += __shfl_xor(v, o, 32);
  return v;
}

// ---------------------------------------------------------------------------
// C[M x 128] = A[M x 128] @ W[128 x 128], bf16 WMMA, f32 accumulate.
// One wave computes a 16-row strip across all 128 columns (8 accumulators).
// Requires M % 16 == 0 (N_NODES = 100000 = 6250 * 16).
// ---------------------------------------------------------------------------
__global__ __launch_bounds__(128) void gcn_gemm_wmma(
    const float* __restrict__ A, const float* __restrict__ W,
    float* __restrict__ C, int M) {
  // W^T in LDS as bf16 bits: sWt[n*128 + k]  (32 KB)
  __shared__ __align__(32) unsigned short sWt[FEAT * FEAT];
  for (int i = threadIdx.x; i < FEAT * FEAT; i += 128) {
    const int k = i >> 7, n = i & 127;       // coalesced read of row-major W
    sWt[n * FEAT + k] = f2bf(W[i]);
  }
  __syncthreads();

  const int wave = threadIdx.x >> 5;
  const int lane = threadIdx.x & 31;
  const int m0 = (blockIdx.x * 4 + wave) * 16;
  if (m0 >= M) return;                       // wave-uniform: EXEC all-ones below
  const int half = lane >> 4;                // K sub-block select (contiguous: k = 16*half + s)
  const int lr   = lane & 15;                // row within tile / column within tile

  v8f acc[8];
#pragma unroll
  for (int t = 0; t < 8; ++t)
#pragma unroll
    for (int i = 0; i < 8; ++i) acc[t][i] = 0.0f;

  const float* arow = A + (size_t)(m0 + lr) * FEAT;

#pragma unroll
  for (int k0 = 0; k0 < FEAT; k0 += 32) {
    // --- A fragment: 16-bit A 16x32 layout ---
    // lane<16: K = k0..k0+15; lane>=16: K = k0+16..k0+31 (matches B).
    const float4 r0 = *(const float4*)(arow + k0 + half * 16);
    const float4 r1 = *(const float4*)(arow + k0 + half * 16 + 4);
    const float4 r2 = *(const float4*)(arow + k0 + half * 16 + 8);
    const float4 r3 = *(const float4*)(arow + k0 + half * 16 + 12);
    v16s av;
    av[0]  = (short)f2bf(r0.x); av[1]  = (short)f2bf(r0.y);
    av[2]  = (short)f2bf(r0.z); av[3]  = (short)f2bf(r0.w);
    av[4]  = (short)f2bf(r1.x); av[5]  = (short)f2bf(r1.y);
    av[6]  = (short)f2bf(r1.z); av[7]  = (short)f2bf(r1.w);
    av[8]  = (short)f2bf(r2.x); av[9]  = (short)f2bf(r2.y);
    av[10] = (short)f2bf(r2.z); av[11] = (short)f2bf(r2.w);
    av[12] = (short)f2bf(r3.x); av[13] = (short)f2bf(r3.y);
    av[14] = (short)f2bf(r3.z); av[15] = (short)f2bf(r3.w);
    const v16bf a = __builtin_bit_cast(v16bf, av);

    // --- B fragments from transposed LDS: contiguous 16 bf16 per lane ---
    const int kb = k0 + half * 16;           // lanes 0-15: K lo-half, 16-31: hi
#pragma unroll
    for (int nt = 0; nt < 8; ++nt) {
      const int n = nt * 16 + lr;            // column = lane%16
      const v16s bv = *(const v16s*)(sWt + n * FEAT + kb);  // 32B aligned
      const v16bf b = __builtin_bit_cast(v16bf, bv);
      acc[nt] = __builtin_amdgcn_wmma_f32_16x16x32_bf16(
          /*neg_a=*/false, a, /*neg_b=*/false, b,
          /*c_mod=*/(short)0, acc[nt], /*reuse_a=*/false, /*reuse_b=*/false);
    }
  }

  // C/D layout: VGPR v -> row (v + 8*half) of the tile, col = lane%16
#pragma unroll
  for (int nt = 0; nt < 8; ++nt)
#pragma unroll
    for (int v = 0; v < 8; ++v)
      C[(size_t)(m0 + v + half * 8) * FEAT + nt * 16 + lr] = acc[nt][v];
}

// ---------------------------------------------------------------------------
// Edge scatter: agg[dst] += h[src] * dinv[src]*dinv[dst].  One wave per edge,
// lane l handles features 4l..4l+3 (32*4 = 128). f32 HW atomics (L2 resident).
// ---------------------------------------------------------------------------
__global__ __launch_bounds__(256) void gcn_edge_scatter(
    const float* __restrict__ h, const float* __restrict__ dinv,
    const int* __restrict__ src, const int* __restrict__ dst,
    float* __restrict__ agg, int E) {
  const int e = blockIdx.x * 8 + (threadIdx.x >> 5);
  if (e >= E) return;
  const int lane = threadIdx.x & 31;
  const int s = src[e], d = dst[e];
  const float nrm = dinv[s] * dinv[d];
  const float4 v = ((const float4*)(h + (size_t)s * FEAT))[lane];
  float* o = agg + (size_t)d * FEAT + lane * 4;
  unsafeAtomicAdd(o + 0, v.x * nrm);
  unsafeAtomicAdd(o + 1, v.y * nrm);
  unsafeAtomicAdd(o + 2, v.z * nrm);
  unsafeAtomicAdd(o + 3, v.w * nrm);
}

// ---------------------------------------------------------------------------
// t = agg + h*dinv^2 + bias ; LayerNorm(t)*g + b ; ReLU.  One wave per row.
// ---------------------------------------------------------------------------
__device__ __forceinline__ float4 finish_row(
    const float* __restrict__ agg, const float* __restrict__ hlin,
    const float* __restrict__ dinv, const float* __restrict__ bias,
    const float* __restrict__ lng, const float* __restrict__ lnb,
    int row, int lane) {
  const float di = dinv[row];
  const float sn = di * di;
  const float4 a = ((const float4*)(agg + (size_t)row * FEAT))[lane];
  const float4 h = ((const float4*)(hlin + (size_t)row * FEAT))[lane];
  const float4 b = ((const float4*)bias)[lane];
  float4 t;
  t.x = a.x + h.x * sn + b.x;
  t.y = a.y + h.y * sn + b.y;
  t.z = a.z + h.z * sn + b.z;
  t.w = a.w + h.w * sn + b.w;
  const float mu = wave_sum32(t.x + t.y + t.z + t.w) * (1.0f / FEAT);
  float4 dx;
  dx.x = t.x - mu; dx.y = t.y - mu; dx.z = t.z - mu; dx.w = t.w - mu;
  const float var = wave_sum32(dx.x * dx.x + dx.y * dx.y +
                               dx.z * dx.z + dx.w * dx.w) * (1.0f / FEAT);
  const float is = rsqrtf(var + 1e-5f);
  const float4 g  = ((const float4*)lng)[lane];
  const float4 bb = ((const float4*)lnb)[lane];
  float4 y;
  y.x = fmaxf(dx.x * is * g.x + bb.x, 0.0f);
  y.y = fmaxf(dx.y * is * g.y + bb.y, 0.0f);
  y.z = fmaxf(dx.z * is * g.z + bb.z, 0.0f);
  y.w = fmaxf(dx.w * is * g.w + bb.w, 0.0f);
  return y;
}

__global__ __launch_bounds__(256) void gcn_finish(
    const float* __restrict__ agg, const float* __restrict__ hlin,
    const float* __restrict__ dinv, const float* __restrict__ bias,
    const float* __restrict__ lng, const float* __restrict__ lnb,
    float* __restrict__ out, int N) {
  const int row = blockIdx.x * 8 + (threadIdx.x >> 5);
  if (row >= N) return;
  const int lane = threadIdx.x & 31;
  const float4 y = finish_row(agg, hlin, dinv, bias, lng, lnb, row, lane);
  ((float4*)(out + (size_t)row * FEAT))[lane] = y;   // row-local: safe in-place
}

__global__ __launch_bounds__(256) void gcn_finish_pool(
    const float* __restrict__ agg, const float* __restrict__ hlin,
    const float* __restrict__ dinv, const float* __restrict__ bias,
    const float* __restrict__ lng, const float* __restrict__ lnb,
    const int* __restrict__ batch, float* __restrict__ pooled,
    float* __restrict__ cnt, int N) {
  const int row = blockIdx.x * 8 + (threadIdx.x >> 5);
  if (row >= N) return;
  const int lane = threadIdx.x & 31;
  const float4 y = finish_row(agg, hlin, dinv, bias, lng, lnb, row, lane);
  const int g = batch[row];
  float* p = pooled + (size_t)g * FEAT + lane * 4;
  unsafeAtomicAdd(p + 0, y.x);
  unsafeAtomicAdd(p + 1, y.y);
  unsafeAtomicAdd(p + 2, y.z);
  unsafeAtomicAdd(p + 3, y.w);
  if (lane == 0) unsafeAtomicAdd(cnt + g, 1.0f);
}

// out[g][o] = (pooled[g]/cnt[g]) . lin_w[:,o] + lin_b[o]   (64 x 16)
__global__ __launch_bounds__(256) void gcn_final(
    const float* __restrict__ pooled, const float* __restrict__ cnt,
    const float* __restrict__ lw, const float* __restrict__ lb,
    float* __restrict__ out) {
  const int t = blockIdx.x * 256 + threadIdx.x;
  if (t >= NGRAPH * DOUT) return;
  const int g = t >> 4, o = t & 15;
  float s = 0.0f;
#pragma unroll 8
  for (int f = 0; f < FEAT; ++f) s += pooled[g * FEAT + f] * lw[f * DOUT + o];
  out[t] = s / fmaxf(cnt[g], 1.0f) + lb[o];
}

// -------- small utility kernels (graph-capture safe, no memset) ------------
__global__ void gcn_zero4(float* __restrict__ p, int n) {  // n % 4 == 0
  int i = (blockIdx.x * blockDim.x + threadIdx.x) * 4;
  const int stride = gridDim.x * blockDim.x * 4;
  for (; i + 3 < n; i += stride)
    *(float4*)(p + i) = make_float4(0.f, 0.f, 0.f, 0.f);
}
__global__ void gcn_fill(float* __restrict__ p, float v, int n) {
  const int i = blockIdx.x * blockDim.x + threadIdx.x;
  if (i < n) p[i] = v;
}
__global__ void gcn_deg(const int* __restrict__ dst, float* __restrict__ deg, int E) {
  const int e = blockIdx.x * blockDim.x + threadIdx.x;
  if (e < E) unsafeAtomicAdd(&deg[dst[e]], 1.0f);
}
__global__ void gcn_rsqrt(const float* __restrict__ a, float* __restrict__ o, int n) {
  const int i = blockIdx.x * blockDim.x + threadIdx.x;
  if (i < n) o[i] = rsqrtf(a[i]);
}

// ---------------------------------------------------------------------------
extern "C" void kernel_launch(void* const* d_in, const int* in_sizes, int n_in,
                              void* d_out, int out_size, void* d_ws, size_t ws_size,
                              hipStream_t stream) {
  const float* x     = (const float*)d_in[0];
  const int*   ei    = (const int*)d_in[1];
  const int*   batch = (const int*)d_in[2];
  const float* W1    = (const float*)d_in[3];
  const float* b1    = (const float*)d_in[4];
  const float* W2    = (const float*)d_in[5];
  const float* b2    = (const float*)d_in[6];
  const float* lng   = (const float*)d_in[7];
  const float* lnb   = (const float*)d_in[8];
  const float* lw    = (const float*)d_in[9];
  const float* lb    = (const float*)d_in[10];
  float* out = (float*)d_out;

  const int N = in_sizes[0] / FEAT;   // 100000 (divisible by 16)
  const int E = in_sizes[1] / 2;      // 1600000
  const int* src = ei;
  const int* dst = ei + E;

  // Workspace: two NxF ping-pong buffers + deg/dinv + pooled/cnt (~103 MB)
  const size_t NF = (size_t)N * FEAT;
  float* A      = (float*)d_ws;            // h_lin / h_relu buffer
  float* B      = A + NF;                  // agg / h_lin buffer
  float* deg    = B + NF;
  float* dinv   = deg + N;
  float* pooled = dinv + N;                // NGRAPH*FEAT, cnt right after

  float* cnt = pooled + NGRAPH * FEAT;
  const int tb = 256;
  const int gemmBlocks = (N / 16 + 3) / 4; // 4 waves (16-row strips) per block

  // Degrees with self-loops: deg = 1 + in-degree; dinv = rsqrt(deg)
  gcn_fill<<<(N + tb - 1) / tb, tb, 0, stream>>>(deg, 1.0f, N);
  gcn_deg<<<(E + tb - 1) / tb, tb, 0, stream>>>(dst, deg, E);
  gcn_rsqrt<<<(N + tb - 1) / tb, tb, 0, stream>>>(deg, dinv, N);

  // Zero scatter target + pooled accumulators (+cnt, contiguous)
  gcn_zero4<<<2048, tb, 0, stream>>>(B, (int)NF);
  gcn_zero4<<<1, tb, 0, stream>>>(pooled, NGRAPH * FEAT + NGRAPH);

  // ---- Layer 1 ----
  gcn_gemm_wmma<<<gemmBlocks, 128, 0, stream>>>(x, W1, A, N);          // A = x@W1
  gcn_edge_scatter<<<(E + 7) / 8, tb, 0, stream>>>(A, dinv, src, dst, B, E);
  gcn_finish<<<(N + 7) / 8, tb, 0, stream>>>(B, A, dinv, b1, lng, lnb, A, N);

  // ---- Layer 2 ----
  gcn_gemm_wmma<<<gemmBlocks, 128, 0, stream>>>(A, W2, B, N);          // B = h@W2
  gcn_zero4<<<2048, tb, 0, stream>>>(A, (int)NF);                      // A = agg2
  gcn_edge_scatter<<<(E + 7) / 8, tb, 0, stream>>>(B, dinv, src, dst, A, E);
  gcn_finish_pool<<<(N + 7) / 8, tb, 0, stream>>>(A, B, dinv, b2, lng, lnb,
                                                  batch, pooled, cnt, N);

  // ---- Readout ----
  gcn_final<<<(NGRAPH * DOUT + tb - 1) / tb, tb, 0, stream>>>(pooled, cnt, lw, lb, out);
}
